// GraphAttentionLayer_30039001268354
// MI455X (gfx1250) — matrix operand order repaired and checked by reading, not compiled, not run on hardware
//
#include <hip/hip_runtime.h>

typedef __attribute__((ext_vector_type(16))) _Float16 v16h;
typedef __attribute__((ext_vector_type(8)))  float    v8f;
typedef __attribute__((ext_vector_type(4)))  unsigned int u32x4;
typedef __attribute__((ext_vector_type(8)))  int      i32x8;
typedef __attribute__((ext_vector_type(4)))  int      i32x4;

#define NHEAD 4
#define BATCH 2
#define NN    4096
#define FIN   64
#define FOUT  32
#define MTILE 256
#define VPAD  8   // V row stride = 264 halves = 528B -> conflict-free b128 LDS reads
#define USE_TDM 1 // stage V tiles via tensor_load_to_lds (TDM) instead of manual copy

// ---- workspace layout (bytes) ----
constexpr size_t SZ_VT   = (size_t)NHEAD*BATCH*FOUT*NN*2;   // 2 MB, f16 V^T [hb][o][n]
constexpr size_t SZ_S    = (size_t)NHEAD*BATCH*NN*4;        // 128 KB each
constexpr size_t OFF_VT   = 0;
constexpr size_t OFF_SSRC = OFF_VT   + SZ_VT;
constexpr size_t OFF_SDST = OFF_SSRC + SZ_S;
constexpr size_t OFF_A1   = OFF_SDST + SZ_S;
constexpr size_t OFF_A2   = OFF_A1   + SZ_S;
constexpr size_t OFF_THR  = OFF_A2   + SZ_S;
constexpr size_t OFF_B1   = OFF_THR  + SZ_S;
constexpr size_t OFF_B2   = OFF_B1   + SZ_S;
constexpr size_t OFF_DMAX = OFF_B2   + SZ_S;
constexpr size_t OFF_OUTS = OFF_DMAX + 256;                 // f32 [hb][n][32], 4 MB

// =====================================================================
// K1: h_prime = h @ w  (WMMA f16), plus s_src/s_dst rank-1 reductions,
//     stores V^T as f16 [hb][o][n].
// =====================================================================
__global__ __launch_bounds__(256) void k_project(
    const float* __restrict__ hsrc,   // [B][N][64]
    const float* __restrict__ wsrc,   // [H][64][32]
    const float* __restrict__ a_src,  // [H][32]
    const float* __restrict__ a_dst,  // [H][32]
    _Float16* __restrict__ vt,        // [H*B][32][N]
    float* __restrict__ s_src,        // [H*B][N]
    float* __restrict__ s_dst)
{
  const int tid  = threadIdx.x;
  const int wave = tid >> 5;
  const int lane = tid & 31;
  const int nblk = blockIdx.x & 31;
  const int hb   = blockIdx.x >> 5;
  const int hh   = hb / BATCH;
  const int bb   = hb % BATCH;

  const int n0    = nblk * 128 + wave * 16;
  const int row   = lane & 15;       // A: row M; B: col N; C: col N
  const int khalf = lane >> 4;
  const int kb    = khalf * 8;

  const float* hrow = hsrc + ((size_t)bb * NN + (n0 + row)) * FIN;

  v16h a0, a1v;
#pragma unroll
  for (int i = 0; i < 16; ++i) {
    int k = kb + i + (i >= 8 ? 8 : 0);
    a0[i]  = (_Float16)hrow[k];
    a1v[i] = (_Float16)hrow[32 + k];
  }

  const float* wb = wsrc + (size_t)hh * FIN * FOUT;
  v8f acc0 = {}, acc1 = {};
#pragma unroll
  for (int kc = 0; kc < 2; ++kc) {
    v16h av = (kc == 0) ? a0 : a1v;
#pragma unroll
    for (int c = 0; c < 2; ++c) {
      v16h bt;
#pragma unroll
      for (int i = 0; i < 16; ++i) {
        int f = 32 * kc + kb + i + (i >= 8 ? 8 : 0);
        bt[i] = (_Float16)wb[f * FOUT + 16 * c + row];
      }
      if (c == 0)
        acc0 = __builtin_amdgcn_wmma_f32_16x16x32_f16(false, av, false, bt, (short)0, acc0, false, false);
      else
        acc1 = __builtin_amdgcn_wmma_f32_16x16x32_f16(false, av, false, bt, (short)0, acc1, false, false);
    }
  }

  const float as_lo = a_src[hh * FOUT + row],      ad_lo = a_dst[hh * FOUT + row];
  const float as_hi = a_src[hh * FOUT + 16 + row], ad_hi = a_dst[hh * FOUT + 16 + row];

#pragma unroll
  for (int r = 0; r < 8; ++r) {
    float ps = acc0[r] * as_lo + acc1[r] * as_hi;
    float pd = acc0[r] * ad_lo + acc1[r] * ad_hi;
#pragma unroll
    for (int m = 1; m < 16; m <<= 1) {
      ps += __shfl_xor(ps, m, 32);
      pd += __shfl_xor(pd, m, 32);
    }
    const int nrow = n0 + r + 8 * khalf;
    if (row == 0) {
      s_src[(size_t)hb * NN + nrow] = ps;
      s_dst[(size_t)hb * NN + nrow] = pd;
    }
    vt[((size_t)hb * FOUT + row)      * NN + nrow] = (_Float16)acc0[r];
    vt[((size_t)hb * FOUT + 16 + row) * NN + nrow] = (_Float16)acc1[r];
  }
}

// =====================================================================
// K2: per (h,b): Dmax = max_m s_dst; factor tables
// =====================================================================
__global__ __launch_bounds__(256) void k_stats(
    const float* __restrict__ s_src, const float* __restrict__ s_dst,
    float* __restrict__ a1, float* __restrict__ a2, float* __restrict__ thr,
    float* __restrict__ b1, float* __restrict__ b2, float* __restrict__ dmax_g)
{
  __shared__ float red[8];
  __shared__ float dshare;
  const int hb  = blockIdx.x;
  const int tid = threadIdx.x;
  const float* sd = s_dst + (size_t)hb * NN;
  const float* ss = s_src + (size_t)hb * NN;

  float m = -3.4e38f;
  for (int j = tid; j < NN; j += 256) m = fmaxf(m, sd[j]);
#pragma unroll
  for (int k = 1; k < 32; k <<= 1) m = fmaxf(m, __shfl_xor(m, k, 32));
  if ((tid & 31) == 0) red[tid >> 5] = m;
  __syncthreads();
  if (tid == 0) {
    float mm = red[0];
    for (int i = 1; i < 8; ++i) mm = fmaxf(mm, red[i]);
    dshare = mm;
    dmax_g[hb] = mm;
  }
  __syncthreads();
  const float D = dshare;

  for (int j = tid; j < NN; j += 256) {
    const size_t o = (size_t)hb * NN + j;
    const float s = ss[j];
    const float z = s + D;
    const float rmax = (z >= 0.f) ? z : 0.2f * z;        // exact row max (leaky monotone)
    a1[o]  = __expf(z - rmax);
    a2[o]  = __expf(0.2f * z - rmax);
    thr[o] = __expf(-s - D);
    const float d = sd[j];
    b1[o] = __expf(d - D);
    b2[o] = __expf(0.2f * (d - D));
  }
}

// =====================================================================
// K3: fused attention: out[n,:] = (P @ V) / rowsum(P)
//     V tiles staged via TDM (tensor_load_to_lds), 3 WMMA per K-chunk
// =====================================================================
__global__ __launch_bounds__(256) void k_attn(
    const _Float16* __restrict__ vt,                     // [H*B][32][N] f16
    const float* __restrict__ a1g, const float* __restrict__ a2g,
    const float* __restrict__ thrg,
    const float* __restrict__ b1g, const float* __restrict__ b2g,
    float* __restrict__ outs)                            // [H*B][N][32] f32
{
  __shared__ float    b1_l[NN];                          // 16 KB
  __shared__ float    b2_l[NN];                          // 16 KB
  __shared__ _Float16 vt_l[FOUT][MTILE + VPAD];          // 16.5 KB (row stride 528 B)

  const int tid  = threadIdx.x;
  const int wave = tid >> 5;
  const int lane = tid & 31;
  const int nblk = blockIdx.x & 31;
  const int hb   = blockIdx.x >> 5;
  const size_t base = (size_t)hb * NN;

  for (int j = tid; j < NN; j += 256) {                  // stage factor tables once
    b1_l[j] = b1g[base + j];
    b2_l[j] = b2g[base + j];
  }

  const int n0    = nblk * 128 + wave * 16;
  const int row   = lane & 15;
  const int khalf = lane >> 4;
  const int kb    = khalf * 8;

  const float A1 = a1g[base + n0 + row];
  const float A2 = a2g[base + n0 + row];
  const float TH = thrg[base + n0 + row];

  v16h ones;
#pragma unroll
  for (int i = 0; i < 16; ++i) ones[i] = (_Float16)1.0f;

  v8f acc0 = {}, acc1 = {}, dacc = {};

#if USE_TDM
  // ---- TDM descriptor pieces (wave-uniform -> SGPRs) ----
  // D# group1: data_size=2B(1), pad_enable, pad_interval=6 (512B), pad_amount=3 (16B)
  //            tensor_dim0=4096, tensor_dim1=32, tile_dim0=256, tile_dim1=32,
  //            tensor_dim0_stride=4096
  i32x8 g1;
  g1[0] = (int)((1u << 16) | (1u << 20) | (6u << 22) | (3u << 25));
  g1[1] = (int)((unsigned)NN << 16);        // tensor_dim0[15:0] in [31:16]
  g1[2] = (int)((unsigned)FOUT << 16);      // tensor_dim1[15:0] in [31:16]
  g1[3] = (int)((unsigned)MTILE << 16);     // tile_dim0 in [31:16]
  g1[4] = (int)FOUT;                        // tile_dim1 in [15:0]
  g1[5] = (int)NN;                          // tensor_dim0_stride[31:0]
  g1[6] = 0;
  g1[7] = 0;
  const i32x4 gz4 = {0, 0, 0, 0};           // groups 2/3 unused (2D tensor)
  const i32x8 gz8 = {0, 0, 0, 0, 0, 0, 0, 0};
  const unsigned lds_off = (unsigned)(uintptr_t)(&vt_l[0][0]);
  const unsigned long long gbase =
      (unsigned long long)(uintptr_t)(vt + (size_t)hb * FOUT * NN);
#endif

  for (int mo = 0; mo < NN / MTILE; ++mo) {
    __syncthreads();                                     // previous tile consumed
#if USE_TDM
    if (wave == 0) {
      const unsigned long long ga = gbase + (unsigned long long)mo * (MTILE * 2);
      u32x4 g0;
      g0[0] = 1u;                                        // count=1, user descriptor
      g0[1] = lds_off;                                   // lds_addr
      g0[2] = (unsigned)ga;                              // global_addr[31:0]
      g0[3] = (unsigned)(ga >> 32) | (2u << 30);         // global_addr[56:32] | type=2
      __builtin_amdgcn_tensor_load_to_lds(g0, g1, gz4, gz4, gz8, 0);
      __builtin_amdgcn_s_wait_tensorcnt(0);
    }
#else
    {                                                    // manual stage: 32 x MTILE halves
      const int o   = tid >> 3;
      const int seg = tid & 7;
      const uint4* s4 = reinterpret_cast<const uint4*>(
          vt + ((size_t)hb * FOUT + o) * NN + mo * MTILE + seg * 32);
      uint4* d4 = reinterpret_cast<uint4*>(&vt_l[o][seg * 32]);
      d4[0] = s4[0]; d4[1] = s4[1]; d4[2] = s4[2]; d4[3] = s4[3];
    }
#endif
    __syncthreads();

#pragma unroll 2
    for (int kc = 0; kc < MTILE / 32; ++kc) {
      const int m0 = mo * MTILE + kc * 32;
      // ---- build P tile (A-matrix, 16x32 f16): no exp in inner loop ----
      v16h ah;
#pragma unroll
      for (int i = 0; i < 16; ++i) {
        const int mi = m0 + khalf * 16 + i;
        const float bv1 = b1_l[mi];
        const float bv2 = b2_l[mi];
        const float p = (bv1 >= TH) ? (A1 * bv1) : (A2 * bv2);
        ah[i] = (_Float16)p;
      }
      // ---- B tiles from LDS (32B contiguous per lane) ----
      const int mm = kc * 32 + khalf * 16;
      union { v16h v; uint4 q[2]; } u0, u1;
      {
        const uint4* p0 = reinterpret_cast<const uint4*>(&vt_l[row][mm]);
        u0.q[0] = p0[0]; u0.q[1] = p0[1];
        const uint4* p1 = reinterpret_cast<const uint4*>(&vt_l[16 + row][mm]);
        u1.q[0] = p1[0]; u1.q[1] = p1[1];
      }
      acc0 = __builtin_amdgcn_wmma_f32_16x16x32_f16(false, ah, false, u0.v, (short)0, acc0, false, false);
      acc1 = __builtin_amdgcn_wmma_f32_16x16x32_f16(false, ah, false, u1.v, (short)0, acc1, false, false);
      dacc = __builtin_amdgcn_wmma_f32_16x16x32_f16(false, ah, false, ones, (short)0, dacc, false, false);
    }
  }

  // epilogue: divide by row-sum (same lane/VGPR as output rows), store f32
#pragma unroll
  for (int r = 0; r < 8; ++r) {
    const float inv = 1.0f / dacc[r];
    const int nrow = n0 + r + 8 * khalf;
    const size_t ob = ((size_t)hb * NN + nrow) * FOUT;
    outs[ob + row]      = acc0[r] * inv;
    outs[ob + 16 + row] = acc1[r] * inv;
  }
}

// =====================================================================
// K4: mean over heads + bias -> d_out [B][N][32] f32
// =====================================================================
__global__ __launch_bounds__(256) void k_combine(
    const float* __restrict__ outs, const float* __restrict__ bias,
    float* __restrict__ out)
{
  const int i = blockIdx.x * 256 + threadIdx.x;
  if (i >= BATCH * NN * FOUT) return;
  const int o = i & 31;
  const int n = (i >> 5) & (NN - 1);
  const int b = i >> 17;
  float s = 0.f;
#pragma unroll
  for (int hh = 0; hh < NHEAD; ++hh)
    s += outs[(((size_t)(hh * BATCH + b) * NN) + n) * FOUT + o];
  out[i] = s * (1.0f / NHEAD) + bias[o];
}

extern "C" void kernel_launch(void* const* d_in, const int* in_sizes, int n_in,
                              void* d_out, int out_size, void* d_ws, size_t ws_size,
                              hipStream_t stream)
{
  (void)in_sizes; (void)n_in; (void)out_size; (void)ws_size;
  const float* h     = (const float*)d_in[0];
  const float* w     = (const float*)d_in[1];
  const float* a_src = (const float*)d_in[2];
  const float* a_dst = (const float*)d_in[3];
  const float* bias  = (const float*)d_in[4];

  char* ws = (char*)d_ws;
  _Float16* vt   = (_Float16*)(ws + OFF_VT);
  float* s_src   = (float*)(ws + OFF_SSRC);
  float* s_dst   = (float*)(ws + OFF_SDST);
  float* a1      = (float*)(ws + OFF_A1);
  float* a2      = (float*)(ws + OFF_A2);
  float* thr     = (float*)(ws + OFF_THR);
  float* b1      = (float*)(ws + OFF_B1);
  float* b2      = (float*)(ws + OFF_B2);
  float* dmax    = (float*)(ws + OFF_DMAX);
  float* outs    = (float*)(ws + OFF_OUTS);

  k_project<<<NHEAD * BATCH * (NN / 128), 256, 0, stream>>>(h, w, a_src, a_dst, vt, s_src, s_dst);
  k_stats  <<<NHEAD * BATCH,              256, 0, stream>>>(s_src, s_dst, a1, a2, thr, b1, b2, dmax);
  k_attn   <<<NHEAD * BATCH * (NN / 128), 256, 0, stream>>>(vt, a1, a2, thr, b1, b2, outs);
  k_combine<<<(BATCH * NN * FOUT) / 256,  256, 0, stream>>>(outs, bias, (float*)d_out);
}
